// CrossAttention_1752346656895
// MI455X (gfx1250) — matrix operand order repaired and checked by reading, not compiled, not run on hardware
//
#include <hip/hip_runtime.h>
#include <hip/hip_bf16.h>

typedef __attribute__((ext_vector_type(16))) _Float16 v16h;
typedef __attribute__((ext_vector_type(8)))  _Float16 v8h;
typedef __attribute__((ext_vector_type(8)))  float    v8f;
typedef __attribute__((ext_vector_type(4)))  int      v4i;

#define B_SZ 64
#define NSEQ 512
#define CDIM 128
#define NH   4
#define HD   32

static __device__ __forceinline__ v8f wmma_f16(v16h a, v16h b, v8f c) {
  // D = A(16x32 f16) * B(32x16 f16) + C(16x16 f32)
  return __builtin_amdgcn_wmma_f32_16x16x32_f16(false, a, false, b, (short)0, c,
                                                false, false);
}

// ---- async global->LDS copy (16 bytes per lane), CDNA5 ASYNCcnt path -------
// Probe: builtin signature takes int4 addrspace(1)* (global src, non-const)
// and (by symmetry) int4 addrspace(3)* (LDS dst), plus imm offset + cpol.
static __device__ __forceinline__ void async_copy16(const void* g, void* l) {
#if __has_builtin(__builtin_amdgcn_global_load_async_to_lds_b128)
  __builtin_amdgcn_global_load_async_to_lds_b128(
      (__attribute__((address_space(1))) v4i*)(void*)g,
      (__attribute__((address_space(3))) v4i*)l, 0, 0);
#else
  unsigned long long ga = (unsigned long long)g;
  unsigned int la = (unsigned int)(unsigned long long)l;  // low 32b = LDS offset
  asm volatile("global_load_async_to_lds_b128 %0, %1, off"
               :: "v"(la), "v"(ga) : "memory");
#endif
}

static __device__ __forceinline__ void wait_async0() {
#if __has_builtin(__builtin_amdgcn_s_wait_asynccnt)
  __builtin_amdgcn_s_wait_asynccnt(0);
#else
  asm volatile("s_wait_asynccnt 0x0" ::: "memory");
#endif
}

// A fragment (16x32, MxK) from row-major f16 [row][k], leading dim ld.
// Lane layout (ISA 7.12.2): row = lane%16, halves 0..7 -> K=8g..8g+7,
// halves 8..15 -> K=16+8g..16+8g+7, g = lane/16.
static __device__ __forceinline__ v16h load_a_f16(const _Float16* base, int ld) {
  int lane = threadIdx.x & 31;
  int g = lane >> 4, row = lane & 15;
  const _Float16* p = base + row * ld;
  v8h lo = *(const v8h*)(p + 8 * g);
  v8h hi = *(const v8h*)(p + 16 + 8 * g);
  v16h r;
#pragma unroll
  for (int i = 0; i < 8; ++i) { r[i] = lo[i]; r[8 + i] = hi[i]; }
  return r;
}

// B fragment (32x16, KxN) where the source is stored [n][k] row-major
// (i.e. the mathematical B transposed), leading dim ld.
// Lane layout: col N = lane%16, halves 0..7 -> K=8g..8g+7,
// halves 8..15 -> K=16+8g..16+8g+7, g = lane/16.
static __device__ __forceinline__ v16h load_b_f16(const _Float16* base, int ld) {
  int lane = threadIdx.x & 31;
  int col = lane & 15, g = lane >> 4;
  const _Float16* p = base + col * ld;
  v8h lo = *(const v8h*)(p + 8 * g);
  v8h hi = *(const v8h*)(p + 16 + 8 * g);
  v16h r;
#pragma unroll
  for (int i = 0; i < 8; ++i) { r[i] = lo[i]; r[8 + i] = hi[i]; }
  return r;
}

static __device__ __forceinline__ v16h load_a_f32(const float* base, int ld) {
  int lane = threadIdx.x & 31;
  int g = lane >> 4, row = lane & 15;
  const float* p = base + row * ld;
  v16h r;
#pragma unroll
  for (int i = 0; i < 8; ++i) {
    r[i]     = (_Float16)p[8 * g + i];
    r[8 + i] = (_Float16)p[16 + 8 * g + i];
  }
  return r;
}

static __device__ __forceinline__ v16h load_b_f32(const float* base, int ld) {
  int lane = threadIdx.x & 31;
  int col = lane & 15, g = lane >> 4;
  const float* p = base + col * ld;
  v16h r;
#pragma unroll
  for (int i = 0; i < 8; ++i) {
    r[i]     = (_Float16)p[8 * g + i];
    r[8 + i] = (_Float16)p[16 + 8 * g + i];
  }
  return r;
}

// ---------------------------------------------------------------- pos MLP ----
static __device__ __forceinline__ void ln_relu8(const float* x, const float* g,
                                                const float* b, float* o) {
  float m = 0.f;
#pragma unroll
  for (int j = 0; j < 8; ++j) m += x[j];
  m *= 0.125f;
  float v = 0.f;
#pragma unroll
  for (int j = 0; j < 8; ++j) { float d = x[j] - m; v += d * d; }
  v *= 0.125f;
  float inv = rsqrtf(v + 1e-5f);
#pragma unroll
  for (int j = 0; j < 8; ++j) {
    float t = (x[j] - m) * inv * g[j] + b[j];
    o[j] = t > 0.f ? t : 0.f;
  }
}

__global__ void pos_mlp_kernel(const int* Hp, const int* Wp, const int* Dp,
                               const float* __restrict__ ppw, const float* __restrict__ ppb,
                               const float* __restrict__ g1, const float* __restrict__ b1,
                               const float* __restrict__ w1, const float* __restrict__ bb1,
                               const float* __restrict__ g2, const float* __restrict__ b2,
                               const float* __restrict__ w2, const float* __restrict__ bb2,
                               const float* __restrict__ g3, const float* __restrict__ b3,
                               const float* __restrict__ w3, const float* __restrict__ bb3,
                               float* __restrict__ pos_tab) {
  int H = Hp[0], W = Wp[0], D = Dp[0];
  int RH = 2 * H - 1, RW = 2 * W - 1, RD = 2 * D - 1;
  int tab = RH * RW * RD;
  int t = blockIdx.x * blockDim.x + threadIdx.x;
  if (t >= tab) return;
  int ih = t / (RW * RD), iw = (t / RD) % RW, id = t % RD;
  float in3[3] = {(float)(ih - (H - 1)), (float)(iw - (W - 1)), (float)(id - (D - 1))};
  float a[8], r[8];
#pragma unroll
  for (int j = 0; j < 8; ++j)
    a[j] = ppb[j] + ppw[j * 3 + 0] * in3[0] + ppw[j * 3 + 1] * in3[1] +
           ppw[j * 3 + 2] * in3[2];
  ln_relu8(a, g1, b1, r);
#pragma unroll
  for (int j = 0; j < 8; ++j) {
    float s = bb1[j];
#pragma unroll
    for (int i = 0; i < 8; ++i) s += w1[j * 8 + i] * r[i];
    a[j] = s;
  }
  ln_relu8(a, g2, b2, r);
#pragma unroll
  for (int j = 0; j < 8; ++j) {
    float s = bb2[j];
#pragma unroll
    for (int i = 0; i < 8; ++i) s += w2[j * 8 + i] * r[i];
    a[j] = s;
  }
  ln_relu8(a, g3, b3, r);
#pragma unroll
  for (int hh = 0; hh < NH; ++hh) {
    float s = bb3[hh];
#pragma unroll
    for (int i = 0; i < 8; ++i) s += w3[hh * 8 + i] * r[i];
    pos_tab[t * NH + hh] = s;
  }
}

// --------------------------------------------------------------- QKV GEMM ----
// q = (x @ Wq^T + bq) * hd^-0.5  -> q_ws [b*h][n][d]   (f16, row-major)
// k = y @ Wk^T + bk              -> k_ws [b*h][n][d]   (f16, row-major)
// v = y @ Wv^T + bv              -> vT_ws [b*h][d][n]  (f16, transposed)
__global__ void __launch_bounds__(256) qkv_kernel(
    const float* __restrict__ x, const float* __restrict__ y,
    const float* __restrict__ qkv_w, const float* __restrict__ qkv_b,
    _Float16* __restrict__ q_ws, _Float16* __restrict__ k_ws,
    _Float16* __restrict__ vT_ws) {
  int wid = blockIdx.x * 8 + (threadIdx.x >> 5);
  int mtile = wid / 24, ctile = wid % 24;   // 2048 x 24 wave tiles
  int lane = threadIdx.x & 31, g = lane >> 4;
  int m0 = mtile * 16, c0 = ctile * 16;
  const float* src = (c0 < CDIM) ? x : y;   // q from x; k,v from y
  v8f acc = {};
#pragma unroll
  for (int kk = 0; kk < CDIM; kk += 32) {
    v16h a = load_a_f32(src + (size_t)m0 * CDIM + kk, CDIM);
    v16h b = load_b_f32(qkv_w + (size_t)c0 * CDIM + kk, CDIM);
    acc = wmma_f16(a, b, acc);
  }
  int c = c0 + (lane & 15);
  float bias = qkv_b[c];
  int mat = c >> 7, cc = c & 127, h = cc >> 5, d = cc & 31;
  const float scale = 0.17677669529663687f;  // 32^-0.5
#pragma unroll
  for (int r = 0; r < 8; ++r) {
    int m = m0 + r + 8 * g;
    int bidx = m >> 9, n = m & 511;
    int bh = bidx * NH + h;
    float v = acc[r] + bias;
    if (mat == 0)
      q_ws[((size_t)bh * NSEQ + n) * HD + d] = (_Float16)(v * scale);
    else if (mat == 1)
      k_ws[((size_t)bh * NSEQ + n) * HD + d] = (_Float16)v;
    else
      vT_ws[((size_t)bh * HD + d) * NSEQ + n] = (_Float16)v;
  }
}

// --------------------------------------------------- flash cross-attention ---
// One block = one (b, h, 128-row q block); 8 waves x 16 q rows.
// K/V tiles (2 KB each per 32-m step) are staged cooperatively into LDS with
// async global->LDS copies (ASYNCcnt) -- one b128 per thread per iteration --
// then consumed by all 8 waves as WMMA B fragments from LDS.
__global__ void __launch_bounds__(256) attn_kernel(
    const _Float16* __restrict__ q_ws, const _Float16* __restrict__ k_ws,
    const _Float16* __restrict__ vT_ws, const float* __restrict__ pos_tab,
    const int* Hp, const int* Wp, const int* Dp, _Float16* __restrict__ o_ws) {
  __shared__ __align__(64) _Float16 ldsP[8][16][32];  // 8 KB, 1 KB per wave
  __shared__ __align__(64) _Float16 ldsK[32][32];     // 2 KB, [m_local][d]
  __shared__ __align__(64) _Float16 ldsV[32][32];     // 2 KB, [d][m_local]
  int H = Hp[0], W = Wp[0], D = Dp[0];
  int WD = W * D, RW = 2 * W - 1, RD = 2 * D - 1;

  int bh = blockIdx.x >> 2, nblk = blockIdx.x & 3;
  int b = bh >> 2, h = bh & 3;
  int tid = threadIdx.x;
  int wv = tid >> 5;
  int lane = tid & 31, g = lane >> 4, mcol = lane & 15;
  int qbase = nblk * 128 + wv * 16;

  const _Float16* qp = q_ws + ((size_t)bh * NSEQ + qbase) * HD;
  const _Float16* kp = k_ws + (size_t)bh * NSEQ * HD;
  const _Float16* vp = vT_ws + (size_t)bh * HD * NSEQ;

  v16h qf = load_a_f16(qp, HD);
  v8f O0 = {}, O1 = {};
  float mrun[8], lrun[8];
  int nh_[8], nw_[8], nd_[8];
#pragma unroll
  for (int r = 0; r < 8; ++r) {
    mrun[r] = -1e30f; lrun[r] = 0.f;
    int n = qbase + r + 8 * g;            // C/D layout: row = r + 8*(lane/16)
    nh_[r] = n / WD; nw_[r] = (n / D) % W; nd_[r] = n % D;
  }

  // per-thread staging addresses: waves 0-3 stage K, waves 4-7 stage V
  int tt = tid & 127;
  int vd = tt >> 2, vc = tt & 3;

  for (int mt = 0; mt < NSEQ; mt += 32) {
    if (tid < 128) {
      // K tile rows mt..mt+31 are contiguous 2 KB: thread covers bytes tt*16..
      const char* gk = (const char*)kp + (size_t)mt * (HD * 2) + tt * 16;
      async_copy16(gk, (char*)&ldsK[0][0] + tt * 16);
    } else {
      // V^T tile: 32 rows (d) of 64 B each, row stride NSEQ*2 bytes
      const char* gv = (const char*)vp + (size_t)vd * (NSEQ * 2) +
                       (size_t)mt * 2 + vc * 16;
      async_copy16(gv, (char*)&ldsV[vd][0] + vc * 16);
    }
    wait_async0();
    __syncthreads();  // all staged K/V visible block-wide

    v16h k0 = load_b_f16(&ldsK[0][0], HD);
    v16h k1 = load_b_f16(&ldsK[16][0], HD);
    v8f z = {};
    v8f S0 = wmma_f16(qf, k0, z);
    v8f S1 = wmma_f16(qf, k1, z);

    // relative-position bias: idx computed analytically, gathered from table
    int m0 = mt + mcol, m1 = mt + 16 + mcol;
    int m0h = m0 / WD, m0w = (m0 / D) % W, m0d = m0 % D;
    int m1h = m1 / WD, m1w = (m1 / D) % W, m1d = m1 % D;
#pragma unroll
    for (int r = 0; r < 8; ++r) {
      int i0 = ((nh_[r] - m0h + H - 1) * RW + (nw_[r] - m0w + W - 1)) * RD +
               (nd_[r] - m0d + D - 1);
      int i1 = ((nh_[r] - m1h + H - 1) * RW + (nw_[r] - m1w + W - 1)) * RD +
               (nd_[r] - m1d + D - 1);
      S0[r] += pos_tab[i0 * NH + h];
      S1[r] += pos_tab[i1 * NH + h];
    }

    // online softmax: each row spans 16 lanes of one half-wave group
#pragma unroll
    for (int r = 0; r < 8; ++r) {
      float t = fmaxf(S0[r], S1[r]);
#pragma unroll
      for (int msk = 1; msk < 16; msk <<= 1) t = fmaxf(t, __shfl_xor(t, msk, 32));
      float nm = fmaxf(mrun[r], t);
      float alpha = __expf(mrun[r] - nm);
      mrun[r] = nm;
      float p0 = __expf(S0[r] - nm), p1 = __expf(S1[r] - nm);
      float rs = p0 + p1;
#pragma unroll
      for (int msk = 1; msk < 16; msk <<= 1) rs += __shfl_xor(rs, msk, 32);
      lrun[r] = lrun[r] * alpha + rs;
      O0[r] *= alpha; O1[r] *= alpha;
      ldsP[wv][r + 8 * g][mcol]      = (_Float16)p0;
      ldsP[wv][r + 8 * g][16 + mcol] = (_Float16)p1;
    }
    __syncthreads();  // D-layout -> A-layout conversion through LDS

    v16h pf  = load_a_f16(&ldsP[wv][0][0], 32);
    v16h vf0 = load_b_f16(&ldsV[0][0], 32);   // d = 0..15
    v16h vf1 = load_b_f16(&ldsV[16][0], 32);  // d = 16..31
    O0 = wmma_f16(pf, vf0, O0);
    O1 = wmma_f16(pf, vf1, O1);
    __syncthreads();  // K/V/P consumed into registers; safe to re-stage
  }

#pragma unroll
  for (int r = 0; r < 8; ++r) {
    float inv = 1.0f / lrun[r];
    int n = qbase + r + 8 * g;
    size_t o = ((size_t)b * NSEQ + n) * CDIM + h * HD;
    o_ws[o + mcol]      = (_Float16)(O0[r] * inv);
    o_ws[o + 16 + mcol] = (_Float16)(O1[r] * inv);
  }
}

// ------------------------------------------------------------ output proj ----
__global__ void __launch_bounds__(256) proj_kernel(
    const _Float16* __restrict__ o_ws, const float* __restrict__ proj_w,
    const float* __restrict__ proj_b, float* __restrict__ out) {
  int wid = blockIdx.x * 8 + (threadIdx.x >> 5);
  int mtile = wid >> 3, ctile = wid & 7;   // 2048 x 8 wave tiles
  int lane = threadIdx.x & 31, g = lane >> 4;
  int m0 = mtile * 16, c0 = ctile * 16;
  v8f acc = {};
#pragma unroll
  for (int kk = 0; kk < CDIM; kk += 32) {
    v16h a = load_a_f16(o_ws + (size_t)m0 * CDIM + kk, CDIM);
    v16h b = load_b_f32(proj_w + (size_t)c0 * CDIM + kk, CDIM);
    acc = wmma_f16(a, b, acc);
  }
  int c = c0 + (lane & 15);
  float bias = proj_b[c];
#pragma unroll
  for (int r = 0; r < 8; ++r) {
    int m = m0 + r + 8 * g;
    out[(size_t)m * CDIM + c] = acc[r] + bias;
  }
}

// ------------------------------------------------------------------ launch ---
extern "C" void kernel_launch(void* const* d_in, const int* in_sizes, int n_in,
                              void* d_out, int out_size, void* d_ws, size_t ws_size,
                              hipStream_t stream) {
  const float* x      = (const float*)d_in[0];
  const float* y      = (const float*)d_in[1];
  const int*   Hp     = (const int*)d_in[2];
  const int*   Wp     = (const int*)d_in[3];
  const int*   Dp     = (const int*)d_in[4];
  const float* qkv_w  = (const float*)d_in[5];
  const float* qkv_b  = (const float*)d_in[6];
  const float* proj_w = (const float*)d_in[7];
  const float* proj_b = (const float*)d_in[8];
  const float* ppw    = (const float*)d_in[9];
  const float* ppb    = (const float*)d_in[10];
  const float* g1     = (const float*)d_in[11];
  const float* b1     = (const float*)d_in[12];
  const float* w1     = (const float*)d_in[13];
  const float* bb1    = (const float*)d_in[14];
  const float* g2     = (const float*)d_in[15];
  const float* b2     = (const float*)d_in[16];
  const float* w2     = (const float*)d_in[17];
  const float* bb2    = (const float*)d_in[18];
  const float* g3     = (const float*)d_in[19];
  const float* b3     = (const float*)d_in[20];
  const float* w3     = (const float*)d_in[21];
  const float* bb3    = (const float*)d_in[22];

  const size_t perMat = (size_t)B_SZ * NH * NSEQ * HD;  // 4M halves = 8 MiB
  _Float16* q_ws  = (_Float16*)d_ws;
  _Float16* k_ws  = q_ws + perMat;
  _Float16* vT_ws = k_ws + perMat;
  _Float16* o_ws  = vT_ws + perMat;
  float* pos_tab  = (float*)(o_ws + (size_t)B_SZ * NSEQ * CDIM);

  pos_mlp_kernel<<<16, 256, 0, stream>>>(Hp, Wp, Dp, ppw, ppb, g1, b1, w1, bb1,
                                         g2, b2, w2, bb2, g3, b3, w3, bb3,
                                         pos_tab);
  // (B*N/16) * (384/16) = 49152 wave tiles, 8 waves/block
  qkv_kernel<<<6144, 256, 0, stream>>>(x, y, qkv_w, qkv_b, q_ws, k_ws, vT_ws);
  // B * h * (N/128) = 1024 blocks
  attn_kernel<<<1024, 256, 0, stream>>>(q_ws, k_ws, vT_ws, pos_tab, Hp, Wp, Dp,
                                        o_ws);
  // (B*N/16) * (128/16) = 16384 wave tiles
  proj_kernel<<<2048, 256, 0, stream>>>(o_ws, proj_w, proj_b, (float*)d_out);
}
